// DriverGeneFewShotModel_6373731467792
// MI455X (gfx1250) — matrix operand order repaired and checked by reading, not compiled, not run on hardware
//
#include <hip/hip_runtime.h>
#include <math.h>

typedef __bf16 bf16;
typedef __attribute__((ext_vector_type(16))) __bf16 v16bf;
typedef __attribute__((ext_vector_type(8)))  float  v8f;
typedef __attribute__((ext_vector_type(4)))  int    v4i;

static inline int divup_i(long a, int b) { return (int)((a + b - 1) / b); }

// ---------------- device helpers ----------------
__device__ __forceinline__ float warp_sum32(float v) {
#pragma unroll
  for (int o = 16; o > 0; o >>= 1) v += __shfl_down(v, o, 32);
  return __shfl(v, 0, 32);
}

// ---------------- packing kernels ----------------
// f32 [rows, ksrc] -> bf16 [rows, kdst], zero-padded
__global__ void dgfs_pack_bf16(const float* __restrict__ src, int ksrc,
                               bf16* __restrict__ dst, int kdst, int rows) {
  long g = (long)blockIdx.x * blockDim.x + threadIdx.x;
  long total = (long)rows * kdst;
  if (g >= total) return;
  int kk = (int)(g % kdst);
  long r = g / kdst;
  float v = (kk < ksrc) ? src[r * ksrc + kk] : 0.0f;
  dst[g] = (bf16)v;
}

// concat(a[rows,ka], b[rows,kb]) -> bf16 [rows, kdst] zero-padded
__global__ void dgfs_pack_concat_bf16(const float* __restrict__ a, int ka,
                                      const float* __restrict__ b, int kb,
                                      bf16* __restrict__ dst, int kdst, int rows) {
  long g = (long)blockIdx.x * blockDim.x + threadIdx.x;
  long total = (long)rows * kdst;
  if (g >= total) return;
  int kk = (int)(g % kdst);
  long r = g / kdst;
  float v = 0.0f;
  if (kk < ka) v = a[r * ka + kk];
  else if (kk < ka + kb) v = b[r * kb + (kk - ka)];
  dst[g] = (bf16)v;
}

// Pack f32 W[Ksrc,Nsrc] (row-major, K x N) into WMMA B-fragment order:
// out[((tileN*ksteps+ks)*32 + lane)*16 + j] = W[ks*32 + (j>>3)*16 + (lane>>4)*8 + (j&7)][tileN*16 + (lane&15)]
__global__ void dgfs_pack_wfrag(const float* __restrict__ W, int Ksrc, int Nsrc,
                                bf16* __restrict__ out, int Kpad, int Npad) {
  int ksteps = Kpad >> 5;
  int total = (Npad / 16) * ksteps * 512;
  int g = blockIdx.x * blockDim.x + threadIdx.x;
  if (g >= total) return;
  int j = g & 15;
  int lane = (g >> 4) & 31;
  int rest = g >> 9;
  int ks = rest % ksteps;
  int tn = rest / ksteps;
  int half = lane >> 4, nn = lane & 15;
  int k = ks * 32 + (j >> 3) * 16 + half * 8 + (j & 7);
  int n = tn * 16 + nn;
  float v = (k < Ksrc && n < Nsrc) ? W[(size_t)k * Nsrc + n] : 0.0f;
  out[g] = (bf16)v;
}

// ---------------- WMMA GEMM: C[M,Nout] = act(A[M,Kpad] @ W + bias) ----------------
// A: bf16 row-major [M, Kpad] (Kpad % 32 == 0), Bfrag: pre-packed fragments.
// One wave (32 threads) per 16x16 output tile.
__global__ void dgfs_wmma_gemm(const bf16* __restrict__ A, const bf16* __restrict__ Bfrag,
                               const float* __restrict__ bias, int nbias,
                               float* __restrict__ C, int ldc,
                               int M, int Kpad, int Nout, int act) {
  const int tileM = blockIdx.x, tileN = blockIdx.y;
  const int lane = threadIdx.x;
  const int r = lane & 15, half = lane >> 4;
  const int ksteps = Kpad >> 5;
  int arow = tileM * 16 + r;
  if (arow >= M) arow = M - 1;
  const bf16* ap = A + (size_t)arow * Kpad;
  const v16bf* Bf = reinterpret_cast<const v16bf*>(Bfrag) + (size_t)tileN * ksteps * 32 + lane;
  v8f acc;
#pragma unroll
  for (int i = 0; i < 8; ++i) acc[i] = 0.0f;
  for (int ks = 0; ks < ksteps; ++ks) {
    union { v4i q[2]; v16bf v; } ua;
    const bf16* a0 = ap + ks * 32 + half * 8;
    ua.q[0] = *reinterpret_cast<const v4i*>(a0);        // K k0..k0+7   (this half)
    ua.q[1] = *reinterpret_cast<const v4i*>(a0 + 16);   // K k0+16..+23 (this half)
    v16bf bfr = Bf[(size_t)ks * 32];
    acc = __builtin_amdgcn_wmma_f32_16x16x32_bf16(false, ua.v, false, bfr,
                                                  (short)0, acc, false, false);
  }
  const int col = tileN * 16 + r;
  float bv = (bias != nullptr && col < nbias) ? bias[col] : 0.0f;
#pragma unroll
  for (int i = 0; i < 8; ++i) {
    int orow = tileM * 16 + half * 8 + i;
    if (orow < M && col < Nout) {
      float v = acc[i] + bv;
      if (act) v = fmaxf(v, 0.0f);
      C[(size_t)orow * ldc + col] = v;
    }
  }
}

// ---------------- graph normalization ----------------
__global__ void dgfs_fill(float* p, float v, long n) {
  long g = (long)blockIdx.x * blockDim.x + threadIdx.x;
  if (g < n) p[g] = v;
}
__global__ void dgfs_deg_accum(const int* __restrict__ rowI, const float* __restrict__ w,
                               float* __restrict__ deg, long e) {
  long g = (long)blockIdx.x * blockDim.x + threadIdx.x;
  if (g >= e) return;
  atomicAdd(&deg[rowI[g]], w[g]);
}
__global__ void dgfs_dis_selfw(const float* __restrict__ deg, float* __restrict__ dis,
                               float* __restrict__ selfw, int n) {
  int g = blockIdx.x * blockDim.x + threadIdx.x;
  if (g >= n) return;
  float d = fmaxf(deg[g], 1e-12f);
  float s = rsqrtf(d);
  dis[g] = s;
  selfw[g] = s * s;
}
__global__ void dgfs_edge_nw(const int* __restrict__ rowI, const int* __restrict__ colI,
                             const float* __restrict__ w, const float* __restrict__ dis,
                             float* __restrict__ nw, long e) {
  long g = (long)blockIdx.x * blockDim.x + threadIdx.x;
  if (g >= e) return;
  nw[g] = dis[rowI[g]] * w[g] * dis[colI[g]];
}

// ---------------- SpMM: out = selfw .* h  then  out[row] += nw * h[col] ----------------
__global__ void dgfs_spmm_self(const float* __restrict__ h, const float* __restrict__ selfw,
                               float* __restrict__ out, int n, int F) {
  long g = (long)blockIdx.x * blockDim.x + threadIdx.x;
  long total = (long)n * F;
  if (g >= total) return;
  int r = (int)(g / F);
  out[g] = selfw[r] * h[g];
}
// one wave per edge; float4 (F=128) or float2 (F=64) per lane
__global__ void dgfs_spmm_edges(const int* __restrict__ rowI, const int* __restrict__ colI,
                                const float* __restrict__ nw, const float* __restrict__ h,
                                float* __restrict__ out, int e, int F) {
  int wid = blockIdx.x * (blockDim.x >> 5) + (threadIdx.x >> 5);
  int lane = threadIdx.x & 31;
  if (wid >= e) return;
  int c = colI[wid];
  int r = rowI[wid];
  float w = nw[wid];
  if (F == 128) {
    float4 v = reinterpret_cast<const float4*>(h + (size_t)c * 128)[lane];
    float* o = out + (size_t)r * 128 + lane * 4;
    atomicAdd(o + 0, w * v.x); atomicAdd(o + 1, w * v.y);
    atomicAdd(o + 2, w * v.z); atomicAdd(o + 3, w * v.w);
  } else { // F == 64
    float2 v = reinterpret_cast<const float2*>(h + (size_t)c * 64)[lane];
    float* o = out + (size_t)r * 64 + lane * 2;
    atomicAdd(o + 0, w * v.x); atomicAdd(o + 1, w * v.y);
  }
}

// ---------------- LayerNorm (F=128), one wave per row ----------------
__global__ void dgfs_layernorm(const float* __restrict__ in, float* __restrict__ out,
                               const float* __restrict__ w, const float* __restrict__ b,
                               int n, int relu_after) {
  int row = blockIdx.x * (blockDim.x >> 5) + (threadIdx.x >> 5);
  int lane = threadIdx.x & 31;
  if (row >= n) return;
  float4 v = reinterpret_cast<const float4*>(in + (size_t)row * 128)[lane];
  float s = warp_sum32(v.x + v.y + v.z + v.w);
  float m = s * (1.0f / 128.0f);
  float dx = v.x - m, dy = v.y - m, dz = v.z - m, dw = v.w - m;
  float q = warp_sum32(dx * dx + dy * dy + dz * dz + dw * dw);
  float inv = rsqrtf(q * (1.0f / 128.0f) + 1e-5f);
  float4 wv = reinterpret_cast<const float4*>(w)[lane];
  float4 bv = reinterpret_cast<const float4*>(b)[lane];
  float4 o;
  o.x = dx * inv * wv.x + bv.x; o.y = dy * inv * wv.y + bv.y;
  o.z = dz * inv * wv.z + bv.z; o.w = dw * inv * wv.w + bv.w;
  if (relu_after) {
    o.x = fmaxf(o.x, 0.f); o.y = fmaxf(o.y, 0.f);
    o.z = fmaxf(o.z, 0.f); o.w = fmaxf(o.w, 0.f);
  }
  reinterpret_cast<float4*>(out + (size_t)row * 128)[lane] = o;
}

// ---------------- softmax over 3 scores + weighted fuse; writes alpha ----------------
__global__ void dgfs_fuse(const float* __restrict__ s0, const float* __restrict__ s1,
                          const float* __restrict__ s2,
                          const float* __restrict__ z0, const float* __restrict__ z1,
                          const float* __restrict__ z2,
                          float* __restrict__ fused, float* __restrict__ alpha, int n) {
  int row = blockIdx.x * (blockDim.x >> 5) + (threadIdx.x >> 5);
  int lane = threadIdx.x & 31;
  if (row >= n) return;
  float a = s0[row], b = s1[row], c = s2[row];
  float mx = fmaxf(a, fmaxf(b, c));
  float e0 = __expf(a - mx), e1 = __expf(b - mx), e2 = __expf(c - mx);
  float inv = 1.0f / (e0 + e1 + e2);
  float a0 = e0 * inv, a1 = e1 * inv, a2 = e2 * inv;
  float4 v0 = reinterpret_cast<const float4*>(z0 + (size_t)row * 128)[lane];
  float4 v1 = reinterpret_cast<const float4*>(z1 + (size_t)row * 128)[lane];
  float4 v2 = reinterpret_cast<const float4*>(z2 + (size_t)row * 128)[lane];
  float4 o;
  o.x = a0 * v0.x + a1 * v1.x + a2 * v2.x;
  o.y = a0 * v0.y + a1 * v1.y + a2 * v2.y;
  o.z = a0 * v0.z + a1 * v1.z + a2 * v2.z;
  o.w = a0 * v0.w + a1 * v1.w + a2 * v2.w;
  reinterpret_cast<float4*>(fused + (size_t)row * 128)[lane] = o;
  if (lane == 0) {
    alpha[(size_t)row * 3 + 0] = a0;
    alpha[(size_t)row * 3 + 1] = a1;
    alpha[(size_t)row * 3 + 2] = a2;
  }
}

// ---------------- top-k prototype (single block, m <= 256, F=128) ----------------
__global__ void dgfs_proto(const float* __restrict__ fused, const int* __restrict__ idx,
                           int m, int k, float* __restrict__ proto) {
  __shared__ float center[128];
  __shared__ float dist[256];
  __shared__ int   sel[256];
  int t = threadIdx.x;
  if (t < 128) {
    float s = 0.f;
    for (int i = 0; i < m; ++i) s += fused[(size_t)idx[i] * 128 + t];
    center[t] = s / (float)m;
  }
  __syncthreads();
  if (t < m) {
    const float* z = fused + (size_t)idx[t] * 128;
    float s = 0.f;
    for (int f = 0; f < 128; ++f) { float d = z[f] - center[f]; s += d * d; }
    dist[t] = s;
  }
  __syncthreads();
  if (t < m) {
    float dv = dist[t];
    int rank = 0;
    for (int j = 0; j < m; ++j) {
      float dj = dist[j];
      if (dj < dv || (dj == dv && j < t)) ++rank;
    }
    sel[t] = (rank < k) ? 1 : 0;
  }
  __syncthreads();
  if (t < 128) {
    float s = 0.f;
    for (int i = 0; i < m; ++i) if (sel[i]) s += fused[(size_t)idx[i] * 128 + t];
    proto[t] = s / (float)k;
  }
}

// ---------------- final logits ----------------
__global__ void dgfs_logits(const float* __restrict__ fused, const float* __restrict__ mlp,
                            const float* __restrict__ ppos, const float* __restrict__ pneg,
                            float* __restrict__ out, int n) {
  int row = blockIdx.x * (blockDim.x >> 5) + (threadIdx.x >> 5);
  int lane = threadIdx.x & 31;
  if (row >= n) return;
  float4 z = reinterpret_cast<const float4*>(fused + (size_t)row * 128)[lane];
  float4 pp = reinterpret_cast<const float4*>(ppos)[lane];
  float4 pn = reinterpret_cast<const float4*>(pneg)[lane];
  float dxp = z.x - pp.x, dyp = z.y - pp.y, dzp = z.z - pp.z, dwp = z.w - pp.w;
  float dxn = z.x - pn.x, dyn = z.y - pn.y, dzn = z.z - pn.z, dwn = z.w - pn.w;
  float dp = warp_sum32(dxp * dxp + dyp * dyp + dzp * dzp + dwp * dwp);
  float dn = warp_sum32(dxn * dxn + dyn * dyn + dzn * dzn + dwn * dwn);
  if (lane == 0)
    out[row] = mlp[row] + 0.25f * (dn - dp) / sqrtf(128.0f);
}

// ---------------- host-side GEMM wrapper ----------------
static void launch_gemm(hipStream_t st, const bf16* A, int Kpad,
                        const float* W, int Ksrc, int Nsrc, const float* bias,
                        bf16* wbuf, float* C, int ldc, int Nout, int M, int act) {
  int Npad = (Nsrc + 15) & ~15;
  int ksteps = Kpad >> 5;
  int total = (Npad / 16) * ksteps * 512;
  dgfs_pack_wfrag<<<divup_i(total, 256), 256, 0, st>>>(W, Ksrc, Nsrc, wbuf, Kpad, Npad);
  dim3 grid(divup_i(M, 16), Npad / 16);
  dgfs_wmma_gemm<<<grid, 32, 0, st>>>(A, wbuf, bias, Nsrc, C, ldc, M, Kpad, Nout, act);
}

// Param leaf indexing (jax pytree: dict keys sorted alphabetically, {'W','b'} -> W,b):
// d_in: 0 x, 1 pos_feat, 2 pos_idx, 3 neg_idx, 4..87 params, 88..93 edges
//  4 cls.l1.W  5 cls.l1.b  6 cls.l2.W  7 cls.l2.b
//  8..21 cons encoder (layout: L0: lin.W, lin.b, ln_b, ln_w, pp1.W, pp1.b, pp2.W, pp2.b
//                              L1: lin.W, lin.b, pp1.W, pp1.b, pp2.W, pp2.b)
//  22 fusion.lin.W 23 fusion.lin.b 24 fusion.ln_b 25 fusion.ln_w
//  26..29 gate.go  30..33 gate.path  34..37 gate.ppi   (l1.W,l1.b,l2.W,l2.b)
//  38 input_proj.W 39 input_proj.b
//  40/41 post.go.W/b  42/43 post.path  44/45 post.ppi
//  46..59 spec.go  60..73 spec.path  74..87 spec.ppi
//  88/89 ei/ew ppi   90/91 path   92/93 go

extern "C" void kernel_launch(void* const* d_in, const int* in_sizes, int n_in,
                              void* d_out, int out_size, void* d_ws, size_t ws_size,
                              hipStream_t stream) {
  if (n_in < 94) return;
  auto F = [&](int i) { return (const float*)d_in[i]; };
  auto I = [&](int i) { return (const int*)d_in[i]; };

  const float* x    = F(0);
  const float* posf = F(1);
  const int* pos_idx = I(2);
  const int* neg_idx = I(3);
  const int m_pos = in_sizes[2], m_neg = in_sizes[3];
  const int N = in_sizes[0] / 256;

  const int spec_base[3] = {74, 60, 46};      // view order: ppi, path, go
  const int gate_base[3] = {34, 30, 26};
  const int postW_i[3]   = {44, 42, 40};
  const int postB_i[3]   = {45, 43, 41};

  long Ev[3]; long Emax = 0;
  for (int v = 0; v < 3; ++v) { Ev[v] = in_sizes[89 + 2 * v]; if (Ev[v] > Emax) Emax = Ev[v]; }

  // ---- workspace carve-out ----
  char* base = (char*)d_ws;
  size_t off = 0;
  auto alloc = [&](size_t bytes) -> void* {
    void* p = base + off;
    off = (off + bytes + 255) & ~(size_t)255;
    return p;
  };
  bf16*  xb    = (bf16*)alloc((size_t)N * 256 * 2);
  bf16*  posb  = (bf16*)alloc((size_t)N * 64 * 2);
  float* x0    = (float*)alloc((size_t)N * 128 * 4);
  float* hbuf  = (float*)alloc((size_t)N * 128 * 4);
  float* hpre  = (float*)alloc((size_t)N * 128 * 4);
  float* hprop = (float*)alloc((size_t)N * 128 * 4);
  bf16*  Abf   = (bf16*)alloc((size_t)N * 160 * 2);
  float* pe1   = (float*)alloc((size_t)N * 16 * 4);
  bf16*  pe1b  = (bf16*)alloc((size_t)N * 32 * 2);
  float* pe    = (float*)alloc((size_t)N * 16 * 4);
  float* zspec = (float*)alloc((size_t)N * 64 * 4);
  float* zcons = (float*)alloc((size_t)N * 64 * 4);
  bf16*  zcatb = (bf16*)alloc((size_t)N * 128 * 2);
  float* zv0   = (float*)alloc((size_t)N * 128 * 4);
  float* zv1   = (float*)alloc((size_t)N * 128 * 4);
  float* zv2   = (float*)alloc((size_t)N * 128 * 4);
  bf16*  zb    = (bf16*)alloc((size_t)N * 128 * 2);
  float* g1    = (float*)alloc((size_t)N * 64 * 4);
  bf16*  g1b   = (bf16*)alloc((size_t)N * 64 * 2);
  float* sc0   = (float*)alloc((size_t)N * 4);
  float* sc1   = (float*)alloc((size_t)N * 4);
  float* sc2   = (float*)alloc((size_t)N * 4);
  float* deg   = (float*)alloc((size_t)N * 4);
  float* dis   = (float*)alloc((size_t)N * 4);
  float* selfw = (float*)alloc((size_t)N * 4);
  float* nw    = (float*)alloc((size_t)Emax * 4);
  float* mlp   = (float*)alloc((size_t)N * 4);
  float* ppos  = (float*)alloc(128 * 4);
  float* pneg  = (float*)alloc(128 * 4);
  bf16*  wbuf  = (bf16*)alloc(256 * 128 * 2 + 1024);
  // reused buffers (views are done when these are live)
  float* fused0 = x0;     // pre-fusion-linear fused features
  float* ftmp   = hprop;  // relu(fusion lin) output
  bf16*  fb     = zcatb;  // bf16 of LN'd fused
  float* c1     = hbuf;   // cls hidden
  bf16*  c1b    = zb;

  float* outLogits = (float*)d_out;
  float* outFused  = (float*)d_out + (size_t)N;
  float* outAlpha  = (float*)d_out + (size_t)N + (size_t)N * 128;

  const hipStream_t st = stream;
  const int TB = 256;

  // ---- constant activations to bf16 ----
  dgfs_pack_bf16<<<divup_i((long)N * 256, TB), TB, 0, st>>>(x, 256, xb, 256, N);
  dgfs_pack_bf16<<<divup_i((long)N * 64, TB), TB, 0, st>>>(posf, 64, posb, 64, N);

  // ---- x0 = relu(x @ Win + b) ----
  launch_gemm(st, xb, 256, F(38), 256, 128, F(39), wbuf, x0, 128, 128, N, 1);

  // encoder param view
  struct EncP {
    const float *W0, *b0, *lnb, *lnw, *p1W0, *p1b0, *p2W0, *p2b0;
    const float *W1, *b1, *p1W1, *p1b1, *p2W1, *p2b1;
  };
  auto encAt = [&](int b) {
    EncP e;
    e.W0 = F(b + 0); e.b0 = F(b + 1); e.lnb = F(b + 2); e.lnw = F(b + 3);
    e.p1W0 = F(b + 4); e.p1b0 = F(b + 5); e.p2W0 = F(b + 6); e.p2b0 = F(b + 7);
    e.W1 = F(b + 8); e.b1 = F(b + 9); e.p1W1 = F(b + 10); e.p1b1 = F(b + 11);
    e.p2W1 = F(b + 12); e.p2b1 = F(b + 13);
    return e;
  };

  // run one 2-layer encoder on the current view's graph
  auto run_encoder = [&](const EncP& e, const int* rowI, const int* colI, long E, float* zout) {
    // layer 0 positional MLP: pe = lin(relu(lin(pos, pp1)), pp2)
    launch_gemm(st, posb, 64, e.p1W0, 64, 16, e.p1b0, wbuf, pe1, 16, 16, N, 1);
    dgfs_pack_bf16<<<divup_i((long)N * 32, TB), TB, 0, st>>>(pe1, 16, pe1b, 32, N);
    launch_gemm(st, pe1b, 32, e.p2W0, 16, 16, e.p2b0, wbuf, pe, 16, 16, N, 0);
    // h = (concat(x0, pe) @ W0 + b0), then propagate
    dgfs_pack_concat_bf16<<<divup_i((long)N * 160, TB), TB, 0, st>>>(x0, 128, pe, 16, Abf, 160, N);
    launch_gemm(st, Abf, 160, e.W0, 144, 128, e.b0, wbuf, hpre, 128, 128, N, 0);
    dgfs_spmm_self<<<divup_i((long)N * 128, TB), TB, 0, st>>>(hpre, selfw, hprop, N, 128);
    dgfs_spmm_edges<<<divup_i(E, TB / 32), TB, 0, st>>>(rowI, colI, nw, hpre, hprop, (int)E, 128);
    dgfs_layernorm<<<divup_i(N, TB / 32), TB, 0, st>>>(hprop, hbuf, e.lnw, e.lnb, N, 1);
    // layer 1
    launch_gemm(st, posb, 64, e.p1W1, 64, 16, e.p1b1, wbuf, pe1, 16, 16, N, 1);
    dgfs_pack_bf16<<<divup_i((long)N * 32, TB), TB, 0, st>>>(pe1, 16, pe1b, 32, N);
    launch_gemm(st, pe1b, 32, e.p2W1, 16, 16, e.p2b1, wbuf, pe, 16, 16, N, 0);
    dgfs_pack_concat_bf16<<<divup_i((long)N * 160, TB), TB, 0, st>>>(hbuf, 128, pe, 16, Abf, 160, N);
    launch_gemm(st, Abf, 160, e.W1, 144, 64, e.b1, wbuf, hpre, 64, 64, N, 0);
    dgfs_spmm_self<<<divup_i((long)N * 64, TB), TB, 0, st>>>(hpre, selfw, zout, N, 64);
    dgfs_spmm_edges<<<divup_i(E, TB / 32), TB, 0, st>>>(rowI, colI, nw, hpre, zout, (int)E, 64);
  };

  EncP consP = encAt(8);
  float* zvs[3] = {zv0, zv1, zv2};
  float* scs[3] = {sc0, sc1, sc2};

  for (int v = 0; v < 3; ++v) {
    const int* ei = I(88 + 2 * v);
    const float* ew = F(89 + 2 * v);
    const int* rowI = ei;
    const int* colI = ei + Ev[v];
    long E = Ev[v];
    // symmetric degree normalization with self loops
    dgfs_fill<<<divup_i(N, TB), TB, 0, st>>>(deg, 1.0f, N);
    dgfs_deg_accum<<<divup_i(E, TB), TB, 0, st>>>(rowI, ew, deg, E);
    dgfs_dis_selfw<<<divup_i(N, TB), TB, 0, st>>>(deg, dis, selfw, N);
    dgfs_edge_nw<<<divup_i(E, TB), TB, 0, st>>>(rowI, colI, ew, dis, nw, E);
    // encoders
    run_encoder(encAt(spec_base[v]), rowI, colI, E, zspec);
    run_encoder(consP, rowI, colI, E, zcons);
    // z = relu(concat(z_spec, z_cons) @ post)
    dgfs_pack_concat_bf16<<<divup_i((long)N * 128, TB), TB, 0, st>>>(zspec, 64, zcons, 64, zcatb, 128, N);
    launch_gemm(st, zcatb, 128, F(postW_i[v]), 128, 128, F(postB_i[v]), wbuf, zvs[v], 128, 128, N, 1);
    // gate score
    dgfs_pack_bf16<<<divup_i((long)N * 128, TB), TB, 0, st>>>(zvs[v], 128, zb, 128, N);
    int gb = gate_base[v];
    launch_gemm(st, zb, 128, F(gb + 0), 128, 64, F(gb + 1), wbuf, g1, 64, 64, N, 1);
    dgfs_pack_bf16<<<divup_i((long)N * 64, TB), TB, 0, st>>>(g1, 64, g1b, 64, N);
    launch_gemm(st, g1b, 64, F(gb + 2), 64, 1, F(gb + 3), wbuf, scs[v], 1, 1, N, 0);
  }

  // ---- softmax over 3 scores + fuse (alpha goes straight to output) ----
  dgfs_fuse<<<divup_i(N, TB / 32), TB, 0, st>>>(sc0, sc1, sc2, zv0, zv1, zv2, fused0, outAlpha, N);

  // ---- fused = ln(relu(fused0 @ Wf + bf)) -> written straight into d_out ----
  dgfs_pack_bf16<<<divup_i((long)N * 128, TB), TB, 0, st>>>(fused0, 128, zb, 128, N);
  launch_gemm(st, zb, 128, F(22), 128, 128, F(23), wbuf, ftmp, 128, 128, N, 1);
  dgfs_layernorm<<<divup_i(N, TB / 32), TB, 0, st>>>(ftmp, outFused, F(25), F(24), N, 0);

  // ---- classifier MLP ----
  dgfs_pack_bf16<<<divup_i((long)N * 128, TB), TB, 0, st>>>(outFused, 128, fb, 128, N);
  launch_gemm(st, fb, 128, F(4), 128, 128, F(5), wbuf, c1, 128, 128, N, 1);
  dgfs_pack_bf16<<<divup_i((long)N * 128, TB), TB, 0, st>>>(c1, 128, c1b, 128, N);
  launch_gemm(st, c1b, 128, F(6), 128, 1, F(7), wbuf, mlp, 1, 1, N, 0);

  // ---- prototypes ----
  int k_pos = (int)((double)m_pos * 0.6); if (k_pos < 8) k_pos = 8; if (k_pos > m_pos) k_pos = m_pos;
  int k_neg = (int)((double)m_neg * 0.6); if (k_neg < 8) k_neg = 8; if (k_neg > m_neg) k_neg = m_neg;
  dgfs_proto<<<1, 256, 0, st>>>(outFused, pos_idx, m_pos, k_pos, ppos);
  dgfs_proto<<<1, 256, 0, st>>>(outFused, neg_idx, m_neg, k_neg, pneg);

  // ---- final logits ----
  dgfs_logits<<<divup_i(N, TB / 32), TB, 0, st>>>(outFused, mlp, ppos, pneg, outLogits, N);

  (void)out_size; (void)ws_size;
}
